// MultiHeadAttentionCausal_2370821947911
// MI455X (gfx1250) — matrix operand order repaired and checked by reading, not compiled, not run on hardware
//
#include <hip/hip_runtime.h>
#include <math.h>

typedef __attribute__((ext_vector_type(16))) __bf16 v16bf;
typedef __attribute__((ext_vector_type(8)))  float  v8f;
typedef int i32x4 __attribute__((vector_size(16)));

#define B_DIM 2
#define T_DIM 2048
#define C_DIM 1024
#define H_NUM 16
#define HD_DIM 64
#define ROWS (B_DIM * T_DIM)          // 4096
#define ATT_SCALE 0.125f              // 1/sqrt(64)

// -------- CDNA5 async global->LDS path (guarded; falls back to ds staging)
#if defined(__gfx1250__) && __has_builtin(__builtin_amdgcn_global_load_async_to_lds_b128)
#define HAVE_ASYNC_LDS 1
#else
#define HAVE_ASYNC_LDS 0
#endif

#if HAVE_ASYNC_LDS
static __device__ __forceinline__ void async_copy16(const void* gsrc, void* ldst) {
  __builtin_amdgcn_global_load_async_to_lds_b128(
      (__attribute__((address_space(1))) i32x4*)gsrc,
      (__attribute__((address_space(3))) i32x4*)ldst, 0, 0);
}
#endif

static __device__ __forceinline__ void wait_async0() {
#if HAVE_ASYNC_LDS
#if __has_builtin(__builtin_amdgcn_s_wait_asynccnt)
  __builtin_amdgcn_s_wait_asynccnt(0);
#else
  asm volatile("s_wait_asynccnt 0" ::: "memory");
#endif
#endif
}

// ---------------------------------------------------------------- WMMA helpers
static __device__ __forceinline__ v8f wmma_bf16(v16bf a, v16bf b, v8f c) {
  return __builtin_amdgcn_wmma_f32_16x16x32_bf16(false, a, false, b,
                                                 (short)0, c, false, false);
}

// A-matrix 16x32 bf16 (ISA 7.12.2): lane m=lane&15, hi=lane>=16.
// elements 0..7 -> K = hi*8+e ; elements 8..15 -> K = 16+hi*8+e.
static __device__ __forceinline__ v16bf load_a_frag(const __bf16* row, int hi) {
  v16bf a;
  const int b0 = hi * 8;
#pragma unroll
  for (int e = 0; e < 8; ++e) a[e] = row[b0 + e];
#pragma unroll
  for (int e = 0; e < 8; ++e) a[8 + e] = row[16 + b0 + e];
  return a;
}

// B-matrix 32x16 bf16: lane n=lane&15; elements 0..7 -> K = hi*8+e ; elements 8..15 -> K = 16+hi*8+e.
static __device__ __forceinline__ v16bf load_b_frag(const __bf16* col, int hi) {
  v16bf bv;
  const int b0 = hi * 8;
#pragma unroll
  for (int e = 0; e < 8; ++e) bv[e] = col[b0 + e];
#pragma unroll
  for (int e = 0; e < 8; ++e) bv[8 + e] = col[16 + b0 + e];
  return bv;
}

// ---------------------------------------------------------------- cast kernel
__global__ void cast_f32_to_bf16(const float* __restrict__ x,
                                 __bf16* __restrict__ y, int n) {
  int i = (blockIdx.x * blockDim.x + threadIdx.x) * 4;
  if (i + 3 < n) {
    float4 v = *(const float4*)(x + i);
    y[i + 0] = (__bf16)v.x;
    y[i + 1] = (__bf16)v.y;
    y[i + 2] = (__bf16)v.z;
    y[i + 3] = (__bf16)v.w;
  }
}

// -------------------------------------------- weight transpose+cast (one-time)
// Wt[z][n][k] = W_z[k][n] as bf16.  grid (C/64, C/64, 4), block 256.
__global__ void transpose_cast_w(const float* __restrict__ Wq,
                                 const float* __restrict__ Wk,
                                 const float* __restrict__ Wv,
                                 const float* __restrict__ Wp,
                                 __bf16* __restrict__ wt) {
  __shared__ __bf16 Ts[64][72];
  const int tid = threadIdx.x;
  const int z   = blockIdx.z;
  const float* W = (z == 0) ? Wq : (z == 1) ? Wk : (z == 2) ? Wv : Wp;
  __bf16* out = wt + (size_t)z * C_DIM * C_DIM;
  const int kb = blockIdx.x * 64;
  const int nb = blockIdx.y * 64;
  const int r  = tid >> 2;            // 0..63
  const int c0 = (tid & 3) * 16;      // 0,16,32,48
  {
    const float* g = W + (size_t)(kb + r) * C_DIM + nb + c0;
#pragma unroll
    for (int j = 0; j < 16; ++j) Ts[r][c0 + j] = (__bf16)g[j];
  }
  __syncthreads();
  {
    __bf16* og = out + (size_t)(nb + r) * C_DIM + kb + c0;
#pragma unroll
    for (int j = 0; j < 16; ++j) og[j] = Ts[c0 + j][r];
  }
}

// ---------------------------------------------------------------- QKV GEMM
// Y = X(4096x1024 bf16) @ W ; Q,K scattered to [B,H,T,HD]; V to [B,H,HD,T].
__global__ void qkv_gemm(const __bf16* __restrict__ xb,
                         const __bf16* __restrict__ wt,   // [3][n][k] slab
                         __bf16* __restrict__ qo,
                         __bf16* __restrict__ ko,
                         __bf16* __restrict__ vt) {
  __shared__ __bf16 As[64][40];   // [row][k]
  __shared__ __bf16 Bs[64][40];   // [n][k]

  const int tid  = threadIdx.x;
  const int lane = tid & 31;
  const int w    = tid >> 5;
  const int hi   = (lane >> 4) & 1;
  const int nl   = lane & 15;
  const int m0   = blockIdx.x * 64;
  const int n0   = blockIdx.y * 64;
  const int z    = blockIdx.z;

  const __bf16* wslab = wt + (size_t)z * C_DIM * C_DIM;

  const int rw = (w & 3) * 16;
  const int cw = (w >> 2) * 32;

  v8f acc0 = {}; v8f acc1 = {};

  const int ar = tid >> 2;            // tile row 0..63
  const int ak = (tid & 3) * 8;       // 8 consecutive k

  for (int k0 = 0; k0 < C_DIM; k0 += 32) {
    __syncthreads();
#if HAVE_ASYNC_LDS
    async_copy16(xb    + (size_t)(m0 + ar) * C_DIM + k0 + ak, &As[ar][ak]);
    async_copy16(wslab + (size_t)(n0 + ar) * C_DIM + k0 + ak, &Bs[ar][ak]);
    wait_async0();
#else
    {
      const __bf16* ga = xb    + (size_t)(m0 + ar) * C_DIM + k0 + ak;
      const __bf16* gb = wslab + (size_t)(n0 + ar) * C_DIM + k0 + ak;
#pragma unroll
      for (int j = 0; j < 8; ++j) { As[ar][ak + j] = ga[j]; Bs[ar][ak + j] = gb[j]; }
    }
#endif
    __syncthreads();

    v16bf a  = load_a_frag(&As[rw + nl][0], hi);
    v16bf b0 = load_b_frag(&Bs[cw + nl][0], hi);
    v16bf b1 = load_b_frag(&Bs[cw + 16 + nl][0], hi);
    acc0 = wmma_bf16(a, b0, acc0);
    acc1 = wmma_bf16(a, b1, acc1);
  }

#pragma unroll
  for (int r = 0; r < 8; ++r) {
    int row = m0 + rw + hi * 8 + r;
    int b   = row >> 11;
    int t   = row & (T_DIM - 1);
    int c0  = n0 + cw + nl;
    int c1  = c0 + 16;
    if (z == 2) {  // V -> [B,H,HD,T]
      size_t o0 = (((size_t)b * H_NUM + (c0 >> 6)) * HD_DIM + (c0 & 63)) * T_DIM + t;
      size_t o1 = (((size_t)b * H_NUM + (c1 >> 6)) * HD_DIM + (c1 & 63)) * T_DIM + t;
      vt[o0] = (__bf16)acc0[r];
      vt[o1] = (__bf16)acc1[r];
    } else {       // Q/K -> [B,H,T,HD]
      __bf16* outp = (z == 0) ? qo : ko;
      size_t o0 = (((size_t)b * H_NUM + (c0 >> 6)) * T_DIM + t) * HD_DIM + (c0 & 63);
      size_t o1 = (((size_t)b * H_NUM + (c1 >> 6)) * T_DIM + t) * HD_DIM + (c1 & 63);
      outp[o0] = (__bf16)acc0[r];
      outp[o1] = (__bf16)acc1[r];
    }
  }
}

// ---------------------------------------------------------------- Flash attention
__global__ void flash_attn(const __bf16* __restrict__ q,
                           const __bf16* __restrict__ k,
                           const __bf16* __restrict__ vt,
                           __bf16* __restrict__ o) {
  __shared__ __bf16 Qs[64][72];       // [qrow][dim]
  __shared__ __bf16 Ks[32][72];       // [key][dim]
  __shared__ __bf16 Vs[64][40];       // [dim][key]
  __shared__ float  Ps[4][16][33];    // per-wave P scratch

  const int tid  = threadIdx.x;       // 0..127
  const int lane = tid & 31;
  const int w    = tid >> 5;
  const int hi   = (lane >> 4) & 1;
  const int nl   = lane & 15;
  const int qt0  = blockIdx.x * 64;
  const int bh   = blockIdx.y;
  const int b    = bh >> 4;
  const int h    = bh & 15;

  const __bf16* qg  = q  + (size_t)bh * T_DIM * HD_DIM;
  const __bf16* kg  = k  + (size_t)bh * T_DIM * HD_DIM;
  const __bf16* vtg = vt + (size_t)bh * HD_DIM * T_DIM;

  // stage Q tile 64x64
#if HAVE_ASYNC_LDS
#pragma unroll
  for (int c = 0; c < 4; ++c) {
    int chunk = tid + c * 128;                 // 0..511 chunks of 8 bf16
    int row = chunk >> 3, off = (chunk & 7) * 8;
    async_copy16(qg + (size_t)(qt0 + row) * HD_DIM + off, &Qs[row][off]);
  }
  wait_async0();
#else
  {
    int row = tid >> 1;
    int d0  = (tid & 1) * 32;
    const __bf16* g = qg + (size_t)(qt0 + row) * HD_DIM + d0;
#pragma unroll
    for (int j = 0; j < 32; ++j) Qs[row][d0 + j] = g[j];
  }
#endif

  v8f oacc[4] = {};
  float mrow[8], lrow[8];
#pragma unroll
  for (int r = 0; r < 8; ++r) { mrow[r] = -1e30f; lrow[r] = 0.0f; }

  const int kcLast = (qt0 + 63) >> 5;
  for (int kc = 0; kc <= kcLast; ++kc) {
    const int k0 = kc * 32;
    if (kc < kcLast) {
      __builtin_prefetch(kg  + (size_t)(k0 + 32 + lane) * HD_DIM, 0, 0);
      __builtin_prefetch(vtg + (size_t)(lane) * T_DIM + k0 + 32, 0, 0);
    }
    __syncthreads();
#if HAVE_ASYNC_LDS
#pragma unroll
    for (int c = 0; c < 2; ++c) {
      int chunk = tid + c * 128;               // 256 chunks: K tile
      int row = chunk >> 3, off = (chunk & 7) * 8;
      async_copy16(kg + (size_t)(k0 + row) * HD_DIM + off, &Ks[row][off]);
    }
#pragma unroll
    for (int c = 0; c < 2; ++c) {
      int chunk = tid + c * 128;               // 256 chunks: V tile (pre-transposed)
      int d = chunk >> 2, off = (chunk & 3) * 8;
      async_copy16(vtg + (size_t)d * T_DIM + k0 + off, &Vs[d][off]);
    }
    wait_async0();
#else
    {
      int key = tid >> 2;
      int d0  = (tid & 3) * 16;
      const __bf16* g = kg + (size_t)(k0 + key) * HD_DIM + d0;
#pragma unroll
      for (int j = 0; j < 16; ++j) Ks[key][d0 + j] = g[j];
    }
    {
      int d   = tid >> 1;
      int off = (tid & 1) * 16;
      const __bf16* g = vtg + (size_t)d * T_DIM + k0 + off;
#pragma unroll
      for (int j = 0; j < 16; ++j) Vs[d][off + j] = g[j];
    }
#endif
    __syncthreads();

    // S(16x32) = Q(16x64) @ K^T : 4 WMMAs
    v8f s0 = {}; v8f s1 = {};
    {
      const __bf16* qrow = &Qs[w * 16 + nl][0];
      v16bf aLo = load_a_frag(qrow, hi);
      v16bf aHi = load_a_frag(qrow + 32, hi);
      v16bf b0l = load_b_frag(&Ks[nl][0], hi);
      v16bf b0h = load_b_frag(&Ks[nl][32], hi);
      v16bf b1l = load_b_frag(&Ks[16 + nl][0], hi);
      v16bf b1h = load_b_frag(&Ks[16 + nl][32], hi);
      s0 = wmma_bf16(aLo, b0l, s0);
      s0 = wmma_bf16(aHi, b0h, s0);
      s1 = wmma_bf16(aLo, b1l, s1);
      s1 = wmma_bf16(aHi, b1h, s1);
    }

    const int qrowBase = qt0 + w * 16 + hi * 8;
#pragma unroll
    for (int r = 0; r < 8; ++r) {
      int qr = qrowBase + r;
      float e0 = s0[r] * ATT_SCALE;
      float e1 = s1[r] * ATT_SCALE;
      if (k0 + nl      > qr) e0 = -1e30f;
      if (k0 + 16 + nl > qr) e1 = -1e30f;
      float t = fmaxf(e0, e1);
      t = fmaxf(t, __shfl_xor(t, 1, 16));
      t = fmaxf(t, __shfl_xor(t, 2, 16));
      t = fmaxf(t, __shfl_xor(t, 4, 16));
      t = fmaxf(t, __shfl_xor(t, 8, 16));
      float mnew  = fmaxf(mrow[r], t);
      float alpha = __expf(mrow[r] - mnew);
      float p0 = __expf(e0 - mnew);
      float p1 = __expf(e1 - mnew);
      float ps = p0 + p1;
      ps += __shfl_xor(ps, 1, 16);
      ps += __shfl_xor(ps, 2, 16);
      ps += __shfl_xor(ps, 4, 16);
      ps += __shfl_xor(ps, 8, 16);
      lrow[r] = lrow[r] * alpha + ps;
      mrow[r] = mnew;
#pragma unroll
      for (int j = 0; j < 4; ++j) oacc[j][r] *= alpha;
      Ps[w][hi * 8 + r][nl]      = p0;
      Ps[w][hi * 8 + r][16 + nl] = p1;
    }
    asm volatile("s_wait_dscnt 0" ::: "memory");

    // O(16x64) += P(16x32) @ V(32x64) : 4 WMMAs
    v16bf pa;
    {
      const float* prow = &Ps[w][nl][0];
      const int b0 = hi * 8;
#pragma unroll
      for (int e = 0; e < 8; ++e) pa[e]     = (__bf16)prow[b0 + e];
#pragma unroll
      for (int e = 0; e < 8; ++e) pa[8 + e] = (__bf16)prow[16 + b0 + e];
    }
#pragma unroll
    for (int j = 0; j < 4; ++j) {
      v16bf vb = load_b_frag(&Vs[j * 16 + nl][0], hi);
      oacc[j] = wmma_bf16(pa, vb, oacc[j]);
    }
  }

#pragma unroll
  for (int r = 0; r < 8; ++r) {
    float inv  = 1.0f / lrow[r];
    int   trow = qt0 + w * 16 + hi * 8 + r;
    size_t base = ((size_t)b * T_DIM + trow) * C_DIM + (size_t)h * HD_DIM;
#pragma unroll
    for (int j = 0; j < 4; ++j)
      o[base + j * 16 + nl] = (__bf16)(oacc[j][r] * inv);
  }
}

// ---------------------------------------------------------------- output projection
__global__ void proj_gemm(const __bf16* __restrict__ ab,
                          const __bf16* __restrict__ wtp,  // Wp transposed bf16
                          const float* __restrict__ bp,
                          float* __restrict__ out) {
  __shared__ __bf16 As[64][40];
  __shared__ __bf16 Bs[64][40];

  const int tid  = threadIdx.x;
  const int lane = tid & 31;
  const int w    = tid >> 5;
  const int hi   = (lane >> 4) & 1;
  const int nl   = lane & 15;
  const int m0   = blockIdx.x * 64;
  const int n0   = blockIdx.y * 64;

  const int rw = (w & 3) * 16;
  const int cw = (w >> 2) * 32;

  v8f acc0 = {}; v8f acc1 = {};

  const int ar = tid >> 2;
  const int ak = (tid & 3) * 8;

  for (int k0 = 0; k0 < C_DIM; k0 += 32) {
    __syncthreads();
#if HAVE_ASYNC_LDS
    async_copy16(ab  + (size_t)(m0 + ar) * C_DIM + k0 + ak, &As[ar][ak]);
    async_copy16(wtp + (size_t)(n0 + ar) * C_DIM + k0 + ak, &Bs[ar][ak]);
    wait_async0();
#else
    {
      const __bf16* ga = ab  + (size_t)(m0 + ar) * C_DIM + k0 + ak;
      const __bf16* gb = wtp + (size_t)(n0 + ar) * C_DIM + k0 + ak;
#pragma unroll
      for (int j = 0; j < 8; ++j) { As[ar][ak + j] = ga[j]; Bs[ar][ak + j] = gb[j]; }
    }
#endif
    __syncthreads();

    v16bf a  = load_a_frag(&As[rw + nl][0], hi);
    v16bf b0 = load_b_frag(&Bs[cw + nl][0], hi);
    v16bf b1 = load_b_frag(&Bs[cw + 16 + nl][0], hi);
    acc0 = wmma_bf16(a, b0, acc0);
    acc1 = wmma_bf16(a, b1, acc1);
  }

#pragma unroll
  for (int r = 0; r < 8; ++r) {
    int row = m0 + rw + hi * 8 + r;
    int c0  = n0 + cw + nl;
    int c1  = c0 + 16;
    out[(size_t)row * C_DIM + c0] = acc0[r] + bp[c0];
    out[(size_t)row * C_DIM + c1] = acc1[r] + bp[c1];
  }
}

// ---------------------------------------------------------------- launcher
extern "C" void kernel_launch(void* const* d_in, const int* in_sizes, int n_in,
                              void* d_out, int out_size, void* d_ws, size_t ws_size,
                              hipStream_t stream) {
  (void)in_sizes; (void)n_in; (void)out_size; (void)ws_size;
  const float* x  = (const float*)d_in[0];
  const float* Wq = (const float*)d_in[1];
  const float* Wk = (const float*)d_in[2];
  const float* Wv = (const float*)d_in[3];
  const float* Wp = (const float*)d_in[4];
  const float* bp = (const float*)d_in[5];
  float* out = (float*)d_out;

  const size_t E  = (size_t)ROWS * C_DIM;            // 4 Mi elements
  const size_t WE = (size_t)C_DIM * C_DIM;           // 1 Mi elements
  char* wsb = (char*)d_ws;
  __bf16* xb = (__bf16*)(wsb);                       //  8 MB
  __bf16* wt = (__bf16*)(wsb + E * 2);               //  8 MB (4 matrices)
  __bf16* qb = (__bf16*)(wsb + E * 2 + 4 * WE * 2);  //  8 MB
  __bf16* kb = (__bf16*)(wsb + 2 * E * 2 + 4 * WE * 2);
  __bf16* vt = (__bf16*)(wsb + 3 * E * 2 + 4 * WE * 2);
  __bf16* ob = (__bf16*)(wsb + 4 * E * 2 + 4 * WE * 2);

  cast_f32_to_bf16<<<(int)(E / 1024), 256, 0, stream>>>(x, xb, (int)E);
  transpose_cast_w<<<dim3(C_DIM / 64, C_DIM / 64, 4), 256, 0, stream>>>(
      Wq, Wk, Wv, Wp, wt);
  qkv_gemm<<<dim3(ROWS / 64, C_DIM / 64, 3), 256, 0, stream>>>(
      xb, wt, qb, kb, vt);
  flash_attn<<<dim3(T_DIM / 64, B_DIM * H_NUM), 128, 0, stream>>>(
      qb, kb, vt, ob);
  proj_gemm<<<dim3(ROWS / 64, C_DIM / 64), 256, 0, stream>>>(
      ob, wt + 3 * WE, bp, out);
}
